// mLSTM_60009283060037
// MI455X (gfx1250) — matrix operand order repaired and checked by reading, not compiled, not run on hardware
//
#include <hip/hip_runtime.h>
#include <hip/hip_bf16.h>

// ---------------------------------------------------------------------------
// mLSTM (xLSTM matrix-memory) for MI455X / gfx1250, wave32 + WMMA + TDM.
// dims: B=8 T=64 N=64 P=64 H=16 DK=DV=64 HID=16, SCALE = 1/8
// ---------------------------------------------------------------------------

#define DEV __device__ __forceinline__

typedef __bf16 bf16_t;
typedef __attribute__((ext_vector_type(16))) __bf16          v16bf;
typedef __attribute__((ext_vector_type(8)))  float           v8f;
typedef __attribute__((ext_vector_type(2)))  float           v2f;
typedef __attribute__((ext_vector_type(8)))  unsigned short  u16x8;
typedef __attribute__((ext_vector_type(16))) unsigned short  u16x16;
typedef __attribute__((ext_vector_type(4)))  unsigned int    u32x4;
typedef __attribute__((ext_vector_type(8)))  int             i32x8;
typedef __attribute__((ext_vector_type(4)))  int             i32x4;

#define B_DIM   8
#define T_DIM   64
#define SCALE_F 0.125f

#define LDA 72   // bf16 row-major stride (144B, 16B aligned rows)
#define LDC 66   // f32 row stride (264B; even-K starts are 8B aligned)

// ---- LDS carving (bytes).  Dynamic LDS only => allocation starts at LDS 0,
// so these offsets are also the hardware LDS byte addresses used by the TDM D#.
#define SOFF_XG1 0          // fp32 staging for x1  (16384 B) <- TDM
#define SOFF_XG2 16384      // fp32 staging for x2  (16384 B) <- TDM
#define SOFF_SC  32768      // float C  (64 x LDC)   16896 B
#define SOFF_PV  49664      // float v  col-major (64 cols x LDC) 16896 B
#define SOFF_SX1 66560      // bf16 x1  (64 x LDA)    9216 B
#define SOFF_SX2 75776      // bf16 x2
#define SOFF_SQ  84992      // bf16 q
#define SOFF_PWQ 94208      // bf16 packed B-frag layout, 4096 el (8192 B)
#define SOFF_PWK 102400
#define SOFF_PWV 110592
#define SOFF_PKT 118784     // bf16 packed k^T
#define SMEM_BYTES 126976

// ---- WMMA fragment loaders (contiguous-k order in A and B, wave32) -------------------
// bf16 A (16x32) from row-major LDS: lane -> row (lane&15); lanes 0-15 carry
// K {0..15}, lanes 16-31 K {16..31}.  Two aligned 16B runs.
DEV v16bf load_a_bf16(const bf16_t* s, int row0, int k0, int lane) {
  const bf16_t* p = s + (row0 + (lane & 15)) * LDA + k0 + ((lane >> 4) << 4);
  union { struct { u16x8 lo, hi; } h; v16bf v; } u;
  u.h.lo = *(const u16x8*)(p);
  u.h.hi = *(const u16x8*)(p + 8);
  return u.v;
}
// bf16 B (32x16) from packed fragment layout pk[(col*4 + (k>>4))*16 + (k&15)]:
// lane -> col (lane&15); lanes 0-15 K=k0..k0+15, lanes 16-31 K=k0+16..k0+31.
// One contiguous 32B run per lane.
DEV v16bf load_b_packed(const bf16_t* pk, int k0, int col0, int lane) {
  int col = col0 + (lane & 15);
  int kc  = (k0 >> 4) + (lane >> 4);
  const bf16_t* p = pk + (col * 4 + kc) * 16;
  u16x16 w = *(const u16x16*)p;
  return __builtin_bit_cast(v16bf, w);
}

// ---- Tensor Data Mover: 1-D tile (nelem fp32) global -> LDS ---------------
DEV void tdm_load_1d(unsigned lds_off, const float* gaddr, unsigned nelem) {
  unsigned long long ga = (unsigned long long)(uintptr_t)gaddr;
  u32x4 g0;
  g0[0] = 1u;                                            // count=1, user mode
  g0[1] = lds_off;                                       // lds_addr (bytes)
  g0[2] = (unsigned)(ga & 0xffffffffu);                  // global_addr lo
  g0[3] = (unsigned)((ga >> 32) & 0x01ffffffu) | 0x80000000u; // addr hi | type=2
  i32x8 g1;
  g1[0] = 0x20000;                   // workgroup_mask=0, data_size=2 (4B)
  g1[1] = (int)(nelem << 16);        // tensor_dim0[15:0] at bits 63:48
  g1[2] = (int)((nelem >> 16) & 0xffffu) | (1 << 16); // dim0 hi | tensor_dim1=1
  g1[3] = (int)(nelem << 16);        // tensor_dim1 hi=0 | tile_dim0 = nelem
  g1[4] = 1;                         // tile_dim1 = 1, tile_dim2 = 0
  g1[5] = (int)nelem;                // tensor_dim0_stride lo
  g1[6] = 0;                         // stride hi, tensor_dim1_stride lo
  g1[7] = 0;
  i32x4 z4 = {};
#if __clang_major__ >= 23
  i32x8 z8 = {};
  __builtin_amdgcn_tensor_load_to_lds(g0, g1, z4, z4, z8, 0);
#else
  __builtin_amdgcn_tensor_load_to_lds(g0, g1, z4, z4, 0);
#endif
}

// ---------------------------------------------------------------------------
// Kernel A: one workgroup per (cell c, batch b, head h).  8 waves / 256 thr.
// ---------------------------------------------------------------------------
__global__ __launch_bounds__(256) void cell_head_kernel(
    const float* __restrict__ feature, const float* __restrict__ base,
    const float* __restrict__ Wq, const float* __restrict__ bq,
    const float* __restrict__ Wk, const float* __restrict__ bk,
    const float* __restrict__ Wv, const float* __restrict__ bv,
    const float* __restrict__ Wout,
    float* __restrict__ Cws, const float* __restrict__ f_state,
    float* __restrict__ h_pre, int t)
{
  extern __shared__ unsigned char smem[];
  float*  xg1 = (float*)(smem + SOFF_XG1);
  float*  xg2 = (float*)(smem + SOFF_XG2);
  float*  sC  = (float*)(smem + SOFF_SC);
  float*  pv  = (float*)(smem + SOFF_PV);
  bf16_t* sx1 = (bf16_t*)(smem + SOFF_SX1);
  bf16_t* sx2 = (bf16_t*)(smem + SOFF_SX2);
  bf16_t* sq  = (bf16_t*)(smem + SOFF_SQ);
  bf16_t* pwq = (bf16_t*)(smem + SOFF_PWQ);
  bf16_t* pwk = (bf16_t*)(smem + SOFF_PWK);
  bf16_t* pwv = (bf16_t*)(smem + SOFF_PWV);
  bf16_t* pkT = (bf16_t*)(smem + SOFF_PKT);

  const int tid  = threadIdx.x;
  const int lane = tid & 31;
  const int wave = tid >> 5;
  const int bid  = blockIdx.x;
  const int c    = bid >> 7;
  const int b    = (bid >> 4) & 7;
  const int h    = bid & 15;
  const int cb   = c * B_DIM + b;
  const int nn_l = lane & 15;
  const int hs   = lane >> 4;

  const float* xf = feature + (size_t)(b * T_DIM + t) * 4096;
  const float* xb = base    + (size_t)(b * T_DIM + t) * 4096;
  const float* x1 = (c == 0) ? xf : xb;
  const float* x2 = (c == 0) ? xb : xf;

  // ---- phase 0: kick off TDM DMAs for x1/x2 (async, TENSORcnt) ----
  if (wave == 0) {
    tdm_load_1d(SOFF_XG1, x1, 4096u);
    tdm_load_1d(SOFF_XG2, x2, 4096u);
  }

  // ---- phase 1a: stage W slices straight into packed B-fragment layout ----
  for (int idx = tid; idx < 4096; idx += 256) {
    int p = idx >> 6, d = idx & 63;                // p = K row, d = col
    int pi = (d * 4 + (p >> 4)) * 16 + (p & 15);
    pwq[pi] = (bf16_t)Wq[p * 1024 + h * 64 + d];
    pwk[pi] = (bf16_t)Wk[p * 1024 + h * 64 + d];
    pwv[pi] = (bf16_t)Wv[p * 1024 + h * 64 + d];
  }
  if (wave == 0) __builtin_amdgcn_s_wait_tensorcnt(0);
  __syncthreads();

  // ---- phase 1b: convert DMA'd fp32 x1/x2 -> bf16 row-major ----
  for (int idx = tid; idx < 4096; idx += 256) {
    int r = idx >> 6, cc = idx & 63;
    sx1[r * LDA + cc] = (bf16_t)xg1[idx];
    sx2[r * LDA + cc] = (bf16_t)xg2[idx];
  }
  __syncthreads();

  // ---- phase 2: q/k/v projections, 48 tiles, 6 per wave (bf16 WMMA) ----
  for (int tt = 0; tt < 6; ++tt) {
    int tix = wave * 6 + tt;            // uniform per wave
    int mat = tix >> 4;                 // 0=q 1=k 2=v
    int t16 = tix & 15;
    int ti = t16 >> 2, tj = t16 & 3;
    const bf16_t* Am = (mat == 0) ? sx1 : sx2;
    const bf16_t* Bm = (mat == 0) ? pwq : ((mat == 1) ? pwk : pwv);
    const float* bias = (mat == 0) ? bq : ((mat == 1) ? bk : bv);
    v8f acc = {};
#pragma unroll
    for (int k0 = 0; k0 < 64; k0 += 32) {
      v16bf av = load_a_bf16(Am, ti * 16, k0, lane);
      v16bf bw = load_b_packed(Bm, k0, tj * 16, lane);
      acc = __builtin_amdgcn_wmma_f32_16x16x32_bf16(false, av, false, bw,
                                                    (short)0, acc, false, false);
    }
    float bia = bias[h * 64 + tj * 16 + nn_l];
    if (mat == 0) {
#pragma unroll
      for (int r = 0; r < 8; ++r) {
        int m = ti * 16 + r + 8 * hs;
        sq[m * LDA + tj * 16 + nn_l] = (bf16_t)(acc[r] + bia);
      }
    } else if (mat == 1) {
      // k value at (n = 16ti+m, d = 16tj+nn_l) -> packed k^T B-operand:
      // col=n, K=d  =>  pkT[(n*4 + tj)*16 + nn_l]
#pragma unroll
      for (int r = 0; r < 8; ++r) {
        int n = ti * 16 + r + 8 * hs;
        pkT[(n * 4 + tj) * 16 + nn_l] = (bf16_t)(acc[r] + bia);
      }
    } else {
      // v col-major for f32 B frags: pv[dv*LDC + n]
#pragma unroll
      for (int r = 0; r < 8; ++r) {
        int n = ti * 16 + r + 8 * hs;
        pv[(tj * 16 + nn_l) * LDC + n] = acc[r] + bia;
      }
    }
  }
  __syncthreads();

  // ---- phase 3: attn = q@k^T, C = f*C + SCALE*attn (2 tiles / wave) ----
  const float fgate = f_state[cb * 16 + h];
  float* Chead = Cws + ((size_t)(cb * 16 + h) << 12);
  for (int tt = 0; tt < 2; ++tt) {
    int t16 = wave * 2 + tt;
    int ti = t16 >> 2, tj = t16 & 3;
    v8f acc = {};
#pragma unroll
    for (int k0 = 0; k0 < 64; k0 += 32) {
      v16bf av = load_a_bf16(sq, ti * 16, k0, lane);
      v16bf bw = load_b_packed(pkT, k0, tj * 16, lane);
      acc = __builtin_amdgcn_wmma_f32_16x16x32_bf16(false, av, false, bw,
                                                    (short)0, acc, false, false);
    }
#pragma unroll
    for (int r = 0; r < 8; ++r) {
      int m = ti * 16 + r + 8 * hs;
      int n = tj * 16 + nn_l;
      float cn = fgate * Chead[m * 64 + n] + SCALE_F * acc[r];
      Chead[m * 64 + n] = cn;           // persistent fp32 state
      sC[m * LDC + n]   = cn;           // for readout this step
    }
  }
  __syncthreads();

  // ---- phase 4: out = C@v (fp32 WMMA 16x16x4), then W_out reduction ----
  for (int tt = 0; tt < 2; ++tt) {
    int t16 = wave * 2 + tt;
    int ti = t16 >> 2, tj = t16 & 3;
    v8f acc = {};
#pragma unroll
    for (int k0 = 0; k0 < 64; k0 += 4) {
      // f32 A 16x4: M=lane&15; lanes 0-15 K={0,1}, lanes 16-31 K={2,3}
      int ak = k0 + 2 * hs;
      v2f a2 = *(const v2f*)&sC[(ti * 16 + nn_l) * LDC + ak];
      v2f b2 = *(const v2f*)&pv[(tj * 16 + nn_l) * LDC + ak];
      acc = __builtin_amdgcn_wmma_f32_16x16x4_f32(false, a2, false, b2,
                                                  (short)0, acc, false, false);
    }
    // partial[hid] += sum over this tile of out * W_out[hid, h, n, dv]
    const float* Wbase = Wout + (size_t)h * 4096
                       + (size_t)(ti * 16 + 8 * hs) * 64 + tj * 16 + nn_l;
    __builtin_prefetch(Wbase, 0, 1);
#pragma unroll 1
    for (int hid = 0; hid < 16; ++hid) {
      const float* Wp = Wbase + (size_t)hid * 65536;
      float p = 0.f;
#pragma unroll
      for (int r = 0; r < 8; ++r) p += acc[r] * Wp[r * 64];
#pragma unroll
      for (int off = 16; off > 0; off >>= 1) p += __shfl_xor(p, off, 32);
      if (lane == 0) atomicAdd(&h_pre[cb * 16 + hid], p);
    }
  }
}

// ---------------------------------------------------------------------------
// Kernel B: per (cell,batch).  h_t = o * (h_pre + b_out), emit output, then
// gates for step t+1: f = exp(g_f), o = sigmoid(g_o).
// ---------------------------------------------------------------------------
__global__ __launch_bounds__(256) void gate_out_kernel(
    const float* __restrict__ feature, const float* __restrict__ base,
    const float* __restrict__ W_ih, const float* __restrict__ W_hh,
    const float* __restrict__ b_gates, const float* __restrict__ b_out,
    float* __restrict__ h_state, float* __restrict__ h_pre,
    float* __restrict__ f_state, float* __restrict__ o_state,
    float* __restrict__ out, int t_emit, int t_next, int do_emit)
{
  __shared__ float lh[16];
  __shared__ float red[256];
  const int tid = threadIdx.x;
  const int c  = blockIdx.x >> 3;
  const int b  = blockIdx.x & 7;
  const int cb = c * B_DIM + b;

  if (do_emit) {
    if (tid < 16) {
      float hv = o_state[cb * 16 + tid] * (h_pre[cb * 16 + tid] + b_out[tid]);
      lh[tid] = hv;
      h_state[cb * 16 + tid] = hv;
      h_pre[cb * 16 + tid] = 0.0f;                       // reset accumulator
      out[((size_t)b * T_DIM + t_emit) * 32 + c * 16 + tid] = hv;
    }
  } else {
    if (tid < 16) lh[tid] = h_state[cb * 16 + tid];      // zeros at init
  }
  __syncthreads();

  if (t_next < T_DIM) {
    const float* x = ((c == 0) ? feature : base)
                   + (size_t)(b * T_DIM + t_next) * 4096;
    for (int j = 16; j < 48; ++j) {       // i-gate is unused by the reference
      float p = 0.f;
      for (int i = tid; i < 4096; i += 256) p += x[i] * W_ih[(size_t)j * 4096 + i];
      red[tid] = p;
      __syncthreads();
      for (int s2 = 128; s2 > 0; s2 >>= 1) {
        if (tid < s2) red[tid] += red[tid + s2];
        __syncthreads();
      }
      if (tid == 0) {
        float g = red[0] + b_gates[j];
        for (int k2 = 0; k2 < 16; ++k2) g += lh[k2] * W_hh[j * 16 + k2];
        if (j < 32) f_state[cb * 16 + (j - 16)] = __expf(g);
        else        o_state[cb * 16 + (j - 32)] = 1.0f / (1.0f + __expf(-g));
      }
      __syncthreads();
    }
  }
}

__global__ void zero_kernel(float* __restrict__ p, size_t n) {
  size_t i = (size_t)blockIdx.x * blockDim.x + threadIdx.x;
  size_t stride = (size_t)gridDim.x * blockDim.x;
  for (; i < n; i += stride) p[i] = 0.0f;
}

// ---------------------------------------------------------------------------
extern "C" void kernel_launch(void* const* d_in, const int* in_sizes, int n_in,
                              void* d_out, int out_size, void* d_ws, size_t ws_size,
                              hipStream_t stream)
{
  const float* feature = (const float*)d_in[0];
  const float* base    = (const float*)d_in[1];
  const float* W_ih    = (const float*)d_in[2];
  const float* W_hh    = (const float*)d_in[3];
  const float* b_gates = (const float*)d_in[4];
  const float* Wq      = (const float*)d_in[5];
  const float* bq      = (const float*)d_in[6];
  const float* Wk      = (const float*)d_in[7];
  const float* bk      = (const float*)d_in[8];
  const float* Wv      = (const float*)d_in[9];
  const float* bv      = (const float*)d_in[10];
  const float* Wout    = (const float*)d_in[11];
  const float* b_out   = (const float*)d_in[12];
  float* out = (float*)d_out;

  // workspace carving (fp32): C state + h + h_pre + f + o
  float* wsf     = (float*)d_ws;
  float* Cws     = wsf;                                  // 2*8*16*64*64
  float* h_state = Cws + (size_t)2 * 8 * 16 * 64 * 64;   // 256
  float* h_pre   = h_state + 256;                        // 256
  float* f_state = h_pre + 256;                          // 256
  float* o_state = f_state + 256;                        // 256

  const size_t zeroN = (size_t)2 * 8 * 16 * 64 * 64 + 4 * 256;
  zero_kernel<<<dim3(1024), dim3(256), 0, stream>>>(wsf, zeroN);

  // gates for t=0 with h=0 (no output emitted)
  gate_out_kernel<<<dim3(16), dim3(256), 0, stream>>>(
      feature, base, W_ih, W_hh, b_gates, b_out,
      h_state, h_pre, f_state, o_state, out, 0, 0, 0);

  for (int t = 0; t < T_DIM; ++t) {
    cell_head_kernel<<<dim3(256), dim3(256), SMEM_BYTES, stream>>>(
        feature, base, Wq, bq, Wk, bk, Wv, bv, Wout,
        Cws, f_state, h_pre, t);
    gate_out_kernel<<<dim3(16), dim3(256), 0, stream>>>(
        feature, base, W_ih, W_hh, b_gates, b_out,
        h_state, h_pre, f_state, o_state, out, t, t + 1, 1);
  }
}
